// NJODE_39376260169840
// MI455X (gfx1250) — compile-verified, matrix operand order checked by reading, not run on hardware
//
#include <hip/hip_runtime.h>
#include <math.h>

typedef __attribute__((ext_vector_type(16))) __bf16 v16bf;
typedef __attribute__((ext_vector_type(8)))  float  v8f;
typedef __attribute__((ext_vector_type(4)))  unsigned int u32x4;

#define Bsz   512
#define Nst   512
#define Din   100
#define Hdim  200
#define NHdim 256
#define DT_   0.01f
#define Dp    112   // D padded to mult of 16
#define Hp    208   // H padded to mult of 16
#define KODE  320   // 302 -> mult of 32
#define KRO   224   // 200 -> mult of 32
#define KENC  128   // 100 -> mult of 32

// d_ws layout: bf16 (ushort) weights, transposed (Wt[n*Kp + k]), then fp32 padded biases
#define OFF_ODE_W1 0
#define OFF_ODE_W2 81920      // 256*320
#define OFF_RO_W1  135168     // +208*256
#define OFF_RO_W2  192512     // +256*224
#define OFF_ENC_W1 221184     // +112*256
#define OFF_ENC_W2 253952     // +256*128
#define WTOTAL     307200     // +208*256
#define BOFF_ODE1 0
#define BOFF_ODE2 256
#define BOFF_RO1  464
#define BOFF_RO2  720
#define BOFF_ENC1 832
#define BOFF_ENC2 1088

__device__ __forceinline__ unsigned short f2bf(float f) {
  unsigned int u = __float_as_uint(f);
  u += 0x7FFFu + ((u >> 16) & 1u);   // round to nearest even
  return (unsigned short)(u >> 16);
}

// Branch-free tanh. CDNA5 has V_TANH_F32; fall back to exp2/rcp chain.
__device__ __forceinline__ float fast_tanh(float x) {
#if __has_builtin(__builtin_amdgcn_tanhf)
  return __builtin_amdgcn_tanhf(x);
#elif __has_builtin(__builtin_amdgcn_exp2f) && __has_builtin(__builtin_amdgcn_rcpf)
  float xc = fminf(9.0f, fmaxf(-9.0f, x));
  float t  = __builtin_amdgcn_exp2f(xc * 2.8853900817779268f);  // e^(2x)
  return (t - 1.0f) * __builtin_amdgcn_rcpf(t + 1.0f);
#else
  return tanhf(x);
#endif
}

__device__ __forceinline__ void wait_async0() {
#if __has_builtin(__builtin_amdgcn_s_wait_asynccnt)
  __builtin_amdgcn_s_wait_asynccnt(0);
#else
  asm volatile("s_wait_asynccnt 0" ::: "memory");
#endif
}

__device__ __forceinline__ unsigned lds_off(const void* p) {
  // generic LDS address = aperture_hi32 | alloc-relative offset in low 32 bits
  return (unsigned)(unsigned long long)p;
}

__device__ __forceinline__ v16bf frag16(const unsigned short* p) {
  union { u32x4 q[2]; v16bf v; } u;
  u.q[0] = *(const u32x4*)(p);        // 16B chunk (K lo half for this lane)
  u.q[1] = *(const u32x4*)(p + 16);   // 16B chunk (K hi half for this lane)
  return u.v;
}

// One 16x16 f32 tile of A[16xKp] * Wt^T, A in LDS (row-major, ld=Kp),
// Wt in global transposed layout (Wt[n*Kp + k]).
// Per-lane fragment addressing follows the CDNA5 16-bit A/B striping:
// lanes 0-15 hold K = [0..7] and [16..23] of row/col (lane&15);
// lanes 16-31 hold K = [8..15] and [24..31].
__device__ __forceinline__ v8f tile_mm(const unsigned short* Alds,
                                       const unsigned short* Wt,
                                       int ntile, int Kp, int lane) {
  const int r  = lane & 15;
  const int hi = (lane >> 4) * 8;
  const unsigned short* arow = Alds + r * Kp + hi;
  const unsigned short* brow = Wt + (size_t)(ntile * 16 + r) * Kp + hi;
  v8f c = {};
  #pragma unroll
  for (int k0 = 0; k0 < Kp; k0 += 32) {
    v16bf a = frag16(arow + k0);
    v16bf b = frag16(brow + k0);
    c = __builtin_amdgcn_wmma_f32_16x16x32_bf16(false, a, false, b,
                                                (short)0, c, false, false);
  }
  return c;
}

// encoder: h_target = pad(x) + MLP(tanh(x)); conditional on mask unless uncond
__device__ __forceinline__ void encoder_pass(
    const unsigned short* encW1, const unsigned short* encW2,
    const float* enc_b1, const float* enc_b2,
    unsigned short* bufA, unsigned short* bufB,
    float* h, const float* xt, const float* mt_s, bool uncond,
    int tid, int lane, int wave)
{
  for (int idx = tid; idx < 16 * KENC; idx += 256) {
    int m = idx / KENC, c = idx % KENC;
    float v = (c < Din) ? fast_tanh(xt[m * Dp + c]) : 0.f;
    bufA[idx] = f2bf(v);
  }
  __syncthreads();
  for (int t = wave; t < 16; t += 8) {
    v8f c = tile_mm(bufA, encW1, t, KENC, lane);
    int n = t * 16 + (lane & 15), mb = (lane >> 4) * 8;
    float bb = enc_b1[n];
    #pragma unroll
    for (int r2 = 0; r2 < 8; ++r2)
      bufB[(mb + r2) * NHdim + n] = f2bf(fast_tanh(c[r2] + bb));
  }
  __syncthreads();
  for (int t = wave; t < 13; t += 8) {
    v8f c = tile_mm(bufB, encW2, t, NHdim, lane);
    int n = t * 16 + (lane & 15), mb = (lane >> 4) * 8;
    float bb = enc_b2[n];
    #pragma unroll
    for (int r2 = 0; r2 < 8; ++r2) {
      int m = mb + r2;
      float hj = ((n < Din) ? xt[m * Dp + n] : 0.f) + c[r2] + bb;
      if (uncond || mt_s[m] > 0.f) h[m * Hp + n] = hj;
    }
  }
  __syncthreads();
}

// readout: Y = h[:, :D] + MLP(tanh(h))
__device__ __forceinline__ void readout_pass(
    const unsigned short* roW1, const unsigned short* roW2,
    const float* ro_b1, const float* ro_b2,
    unsigned short* bufA, unsigned short* bufB,
    const float* h, float* Y, int tid, int lane, int wave)
{
  for (int idx = tid; idx < 16 * KRO; idx += 256) {
    int m = idx / KRO, c = idx % KRO;
    float v = (c < Hdim) ? fast_tanh(h[m * Hp + c]) : 0.f;
    bufA[idx] = f2bf(v);
  }
  __syncthreads();
  for (int t = wave; t < 16; t += 8) {
    v8f c = tile_mm(bufA, roW1, t, KRO, lane);
    int n = t * 16 + (lane & 15), mb = (lane >> 4) * 8;
    float bb = ro_b1[n];
    #pragma unroll
    for (int r2 = 0; r2 < 8; ++r2)
      bufB[(mb + r2) * NHdim + n] = f2bf(fast_tanh(c[r2] + bb));
  }
  __syncthreads();
  if (wave < 7) {
    v8f c = tile_mm(bufB, roW2, wave, NHdim, lane);
    int n = wave * 16 + (lane & 15), mb = (lane >> 4) * 8;
    float bb = ro_b2[n];
    #pragma unroll
    for (int r2 = 0; r2 < 8; ++r2) {
      int m = mb + r2;
      Y[m * Dp + n] = ((n < Din) ? h[m * Hp + n] : 0.f) + c[r2] + bb;
    }
  }
  __syncthreads();
}

__global__ __launch_bounds__(256, 1) void njode_scan(
    const float* __restrict__ X, const float* __restrict__ mask,
    const float* __restrict__ startX, const float* __restrict__ nobs,
    const unsigned short* __restrict__ w, const float* __restrict__ bias,
    float* __restrict__ out)
{
  __shared__ __align__(16) float h[16 * Hp];
  __shared__ __align__(16) float lastX[16 * Dp];
  __shared__ __align__(16) float xt[16 * Dp];
  __shared__ __align__(16) float Ybj[16 * Dp];
  __shared__ __align__(16) float Ynow[16 * Dp];
  __shared__ __align__(16) unsigned short bufA[16 * KODE];
  __shared__ __align__(16) unsigned short bufB[16 * NHdim];
  __shared__ float red[2 * 256];
  __shared__ float tau_s[16], mt_s[16], nobs_s[16];

  const int tid  = threadIdx.x;
  const int lane = tid & 31;
  const int wave = tid >> 5;
  const int rowbase = blockIdx.x * 16;

  const unsigned short* odeW1 = w + OFF_ODE_W1;
  const unsigned short* odeW2 = w + OFF_ODE_W2;
  const unsigned short* roW1  = w + OFF_RO_W1;
  const unsigned short* roW2  = w + OFF_RO_W2;
  const unsigned short* encW1 = w + OFF_ENC_W1;
  const unsigned short* encW2 = w + OFF_ENC_W2;
  const float* ode_b1 = bias + BOFF_ODE1;
  const float* ode_b2 = bias + BOFF_ODE2;
  const float* ro_b1  = bias + BOFF_RO1;
  const float* ro_b2  = bias + BOFF_RO2;
  const float* enc_b1 = bias + BOFF_ENC1;
  const float* enc_b2 = bias + BOFF_ENC2;

  // ---- init: h=0, lastX=xt=startX, tau=0 ----
  for (int idx = tid; idx < 16 * Hp; idx += 256) h[idx] = 0.f;
  for (int idx = tid; idx < 16 * Dp; idx += 256) {
    int m = idx / Dp, c = idx % Dp;
    float v = (c < Din) ? startX[(size_t)(rowbase + m) * Din + c] : 0.f;
    lastX[idx] = v;
    xt[idx]    = v;
  }
  if (tid < 16) {
    tau_s[tid]  = 0.f;
    mt_s[tid]   = 1.f;
    nobs_s[tid] = nobs[rowbase + tid];
  }
  __syncthreads();

  // h0 = enc(start_X), unconditional
  encoder_pass(encW1, encW2, enc_b1, enc_b2, bufA, bufB, h, xt, mt_s, true,
               tid, lane, wave);

  float lloss = 0.f;

  for (int i = 0; i < Nst; ++i) {
    const float t_prev = i * DT_;
    const float t_now  = t_prev + DT_;

    // ---- phase 0: async-load x_t to LDS, build ode_in (bf16), prefetch ----
    // 16 rows x 100 floats = 400 aligned 16B chunks -> GLOBAL_LOAD_ASYNC_TO_LDS_B128
    for (int idx = tid; idx < 400; idx += 256) {
      int m = idx / 25, c4 = idx % 25;
      const float* g = &X[(size_t)(rowbase + m) * Nst * Din + (size_t)i * Din
                          + (size_t)c4 * 4];
      unsigned l = lds_off(&xt[m * Dp + c4 * 4]);
      asm volatile("global_load_async_to_lds_b128 %0, %1, off"
                   :: "v"(l), "v"(g) : "memory");
    }
    for (int idx = tid; idx < 16 * (Dp - Din); idx += 256) {  // zero pad cols
      int m = idx / (Dp - Din), c = Din + idx % (Dp - Din);
      xt[m * Dp + c] = 0.f;
    }
    for (int idx = tid; idx < 16 * KODE; idx += 256) {
      int m = idx / KODE, c = idx % KODE;
      float v;
      if (c < Din)             v = fast_tanh(lastX[m * Dp + c]);
      else if (c < Din + Hdim) v = fast_tanh(h[m * Hp + (c - Din)]);
      else if (c == 300)       v = tau_s[m];
      else if (c == 301)       v = t_prev - tau_s[m];
      else                     v = 0.f;
      bufA[idx] = f2bf(v);
    }
    if (tid < 16) {
      mt_s[tid] = mask[(size_t)(rowbase + tid) * Nst + i];
      if (i + 1 < Nst)
        __builtin_prefetch(
            &X[(size_t)(rowbase + tid) * Nst * Din + (size_t)(i + 1) * Din],
            0, 1);
    }
    wait_async0();
    __syncthreads();

    // ---- ODE layer 1: [16x320]x[320x256] -> tanh -> bufB ----
    for (int t = wave; t < 16; t += 8) {
      v8f c = tile_mm(bufA, odeW1, t, KODE, lane);
      int n = t * 16 + (lane & 15), mb = (lane >> 4) * 8;
      float bb = ode_b1[n];
      #pragma unroll
      for (int r2 = 0; r2 < 8; ++r2)
        bufB[(mb + r2) * NHdim + n] = f2bf(fast_tanh(c[r2] + bb));
    }
    __syncthreads();

    // ---- ODE layer 2: h += DT * ([16x256]x[256x208] + b) ----
    for (int t = wave; t < 13; t += 8) {
      v8f c = tile_mm(bufB, odeW2, t, NHdim, lane);
      int n = t * 16 + (lane & 15), mb = (lane >> 4) * 8;
      float bb = ode_b2[n];
      #pragma unroll
      for (int r2 = 0; r2 < 8; ++r2)
        h[(mb + r2) * Hp + n] += DT_ * (c[r2] + bb);
    }
    __syncthreads();

    // ---- Y_bj = readout(h) (pre-jump) ----
    readout_pass(roW1, roW2, ro_b1, ro_b2, bufA, bufB, h, Ybj, tid, lane, wave);

    // ---- jump: h = enc(x_t) where observed ----
    encoder_pass(encW1, encW2, enc_b1, enc_b2, bufA, bufB, h, xt, mt_s, false,
                 tid, lane, wave);

    // ---- last_X / tau update where observed ----
    for (int idx = tid; idx < 16 * Dp; idx += 256) {
      int m = idx / Dp;
      if (mt_s[m] > 0.f) lastX[idx] = xt[idx];
    }
    if (tid < 16 && mt_s[tid] > 0.f) tau_s[tid] = t_now;
    // (consumers of lastX/tau are next iteration, after later barriers)

    // ---- Y = readout(h) (post-jump) ----
    readout_pass(roW1, roW2, ro_b1, ro_b2, bufA, bufB, h, Ynow, tid, lane, wave);

    // ---- loss: per-row sqrt-sum-squares, 16 threads per row ----
    {
      int row = tid >> 4, sub = tid & 15;
      float s1 = 0.f, s2 = 0.f;
      for (int c = sub; c < Din; c += 16) {
        float yn = Ynow[row * Dp + c];
        float d1 = xt[row * Dp + c] - yn;  s1 += d1 * d1;
        float d2 = Ybj[row * Dp + c] - yn; s2 += d2 * d2;
      }
      red[tid] = s1;
      red[256 + tid] = s2;
      __syncthreads();
      if (sub == 0) {
        float a = 0.f, b = 0.f;
        #pragma unroll
        for (int j = 0; j < 16; ++j) { a += red[row * 16 + j]; b += red[256 + row * 16 + j]; }
        float sq1 = sqrtf(a + 1e-10f), sq2 = sqrtf(b + 1e-10f);
        float inner = sq1 + sq2;         // 2*0.5*sq1 + 2*(1-0.5)*sq2
        inner *= inner;
        lloss += mt_s[row] * inner / nobs_s[row];
      }
      __syncthreads();
    }
  }

  // ---- final reduction: 16 per-row accumulators -> atomicAdd ----
  if ((tid & 15) == 0) red[tid >> 4] = lloss;
  __syncthreads();
  if (tid == 0) {
    float s = 0.f;
    #pragma unroll
    for (int r = 0; r < 16; ++r) s += red[r];
    atomicAdd(out, s * (1.0f / (float)Bsz));
  }
}

// ---- weight prep: fp32 W[KxN] row-major -> bf16 transposed padded Wt[Np x Kp] ----
__global__ void convert_wt(const float* __restrict__ W, unsigned short* __restrict__ dst,
                           int K, int Nn, int Kp, int Np) {
  int idx = blockIdx.x * 256 + threadIdx.x;
  int total = Np * Kp;
  if (idx >= total) return;
  int n = idx / Kp, k = idx % Kp;
  float v = (k < K && n < Nn) ? W[(size_t)k * Nn + n] : 0.f;
  dst[idx] = f2bf(v);
}

__global__ void pad_bias(const float* __restrict__ b, float* __restrict__ dst,
                         int n, int np) {
  int i = blockIdx.x * 256 + threadIdx.x;
  if (i < np) dst[i] = (i < n) ? b[i] : 0.f;
}

__global__ void zero_out(float* o, int n) {
  int i = blockIdx.x * 256 + threadIdx.x;
  if (i < n) o[i] = 0.f;
}

extern "C" void kernel_launch(void* const* d_in, const int* in_sizes, int n_in,
                              void* d_out, int out_size, void* d_ws, size_t ws_size,
                              hipStream_t stream) {
  // setup_inputs() dict order:
  const float* X      = (const float*)d_in[0];
  const float* mask   = (const float*)d_in[1];
  const float* startX = (const float*)d_in[2];
  const float* enc_W1 = (const float*)d_in[3];
  const float* enc_b1 = (const float*)d_in[4];
  const float* enc_W2 = (const float*)d_in[5];
  const float* enc_b2 = (const float*)d_in[6];
  const float* ode_W1 = (const float*)d_in[7];
  const float* ode_b1 = (const float*)d_in[8];
  const float* ode_W2 = (const float*)d_in[9];
  const float* ode_b2 = (const float*)d_in[10];
  const float* ro_W1  = (const float*)d_in[11];
  const float* ro_b1  = (const float*)d_in[12];
  const float* ro_W2  = (const float*)d_in[13];
  const float* ro_b2  = (const float*)d_in[14];
  const float* nobs   = (const float*)d_in[15];
  float* out = (float*)d_out;

  unsigned short* w = (unsigned short*)d_ws;
  float* bias = (float*)(w + WTOTAL);

  #define CVT(src, off, K, Nn, Kp, Np) \
    convert_wt<<<((Np) * (Kp) + 255) / 256, 256, 0, stream>>>(src, w + (off), K, Nn, Kp, Np)
  CVT(ode_W1, OFF_ODE_W1, 302, 256, KODE, 256);
  CVT(ode_W2, OFF_ODE_W2, 256, 200, NHdim, Hp);
  CVT(ro_W1,  OFF_RO_W1,  200, 256, KRO,  256);
  CVT(ro_W2,  OFF_RO_W2,  256, 100, NHdim, Dp);
  CVT(enc_W1, OFF_ENC_W1, 100, 256, KENC, 256);
  CVT(enc_W2, OFF_ENC_W2, 256, 200, NHdim, Hp);
  #undef CVT

  pad_bias<<<1, 256, 0, stream>>>(ode_b1, bias + BOFF_ODE1, 256, 256);
  pad_bias<<<1, 256, 0, stream>>>(ode_b2, bias + BOFF_ODE2, 200, Hp);
  pad_bias<<<1, 256, 0, stream>>>(ro_b1,  bias + BOFF_RO1,  256, 256);
  pad_bias<<<1, 256, 0, stream>>>(ro_b2,  bias + BOFF_RO2,  100, Dp);
  pad_bias<<<1, 256, 0, stream>>>(enc_b1, bias + BOFF_ENC1, 256, 256);
  pad_bias<<<1, 256, 0, stream>>>(enc_b2, bias + BOFF_ENC2, 200, Hp);

  zero_out<<<1, 32, 0, stream>>>(out, out_size);

  njode_scan<<<Bsz / 16, 256, 0, stream>>>(X, mask, startX, nobs, w, bias, out);
}
